// MultiHeadAttention_83734682403111
// MI455X (gfx1250) — matrix operand order repaired and checked by reading, not compiled, not run on hardware
//
#include <hip/hip_runtime.h>

#define B_ 4
#define S_ 1024
#define D_ 768
#define H_ 12
#define DK_ 64

typedef __attribute__((ext_vector_type(16))) _Float16 v16h;
typedef __attribute__((ext_vector_type(8)))  _Float16 v8h;
typedef __attribute__((ext_vector_type(8)))  float    v8f;
typedef __attribute__((ext_vector_type(4)))  float    v4f;
typedef __attribute__((ext_vector_type(4)))  int      v4i;

// Build a 16-half WMMA fragment from two contiguous 8-half vector loads.
__device__ __forceinline__ v16h make16(v8h lo, v8h hi) {
    v16h r;
    #pragma unroll
    for (int i = 0; i < 8; ++i) { r[i] = lo[i]; r[i + 8] = hi[i]; }
    return r;
}

// A-fragment (16x32 f16) from a row-major K-run: runs [K0..K0+7], [K0+16..K0+23],
// K0 = half*8  -> two b128 loads.
__device__ __forceinline__ v16h load_frag_a(const _Float16* base, int half) {
    v8h lo = *(const v8h*)(base + half * 8);
    v8h hi = *(const v8h*)(base + half * 8 + 16);
    return make16(lo, hi);
}

// B-fragment (32x16 f16) from a row-major K-run: runs [K0..K0+7], [K0+16..K0+23],
// K0 = half*8  -> two b128 loads.
__device__ __forceinline__ v16h load_frag_b(const _Float16* base, int half) {
    v8h lo = *(const v8h*)(base + half * 8);
    v8h hi = *(const v8h*)(base + half * 8 + 16);
    return make16(lo, hi);
}

// ---------------------------------------------------------------------------
// Kernel 1: fused 4-way projection GEMM, computed TRANSPOSED:
//   C^T = W^T @ X^T   (A-frag = W columns from WsT, B-frag = X rows from Xs)
// so each lane owns 8 consecutive output columns of one row -> b128 stores.
// Staging: each thread owns 8 consecutive columns -> b128 global loads,
// b128 LDS store for Xs, transposed b16 scatter for WsT.
// block = 256 threads (8 waves), one 64x64 output tile.
// ---------------------------------------------------------------------------
__global__ __launch_bounds__(256)
void proj_kernel(const float* __restrict__ query, const float* __restrict__ key,
                 const float* __restrict__ aspect,
                 const float* __restrict__ Wq,  const float* __restrict__ bq,
                 const float* __restrict__ Wk,  const float* __restrict__ bk,
                 const float* __restrict__ Wqa, const float* __restrict__ bqa,
                 const float* __restrict__ Wka, const float* __restrict__ bka,
                 _Float16* __restrict__ qh,  _Float16* __restrict__ kh,
                 _Float16* __restrict__ qah, _Float16* __restrict__ kah)
{
    __shared__ _Float16 Xs[64][72];    // [row][k]  stride 144B = 9*16B aligned
    __shared__ _Float16 WsT[64][72];   // [col][k]  transposed at fill time

    const int tid  = threadIdx.x;
    const int lane = tid & 31;
    const int wave = tid >> 5;
    const int half = lane >> 4;
    const int ln16 = lane & 15;
    const int bx = blockIdx.x;    // column tile (0..11)
    const int by = blockIdx.y;    // row tile    (0..63)
    const int p  = blockIdx.z;    // projection

    const float* X; const float* W; const float* bias; _Float16* out;
    if (p == 0)      { X = query;  W = Wq;  bias = bq;  out = qh;  }
    else if (p == 1) { X = key;    W = Wk;  bias = bk;  out = kh;  }
    else if (p == 2) { X = aspect; W = Wqa; bias = bqa; out = qah; }
    else             { X = aspect; W = Wka; bias = bka; out = kah; }

    v8f acc[2] = {{}, {}};
    const int st0 = wave * 2;     // each wave owns two 16x16 subtiles

    for (int kk = 0; kk < D_; kk += 64) {
        // 512 groups of 8 consecutive columns; 2 groups per thread.
        #pragma unroll
        for (int g = 0; g < 2; ++g) {
            const int idx = tid + g * 256;       // 0..511
            const int r  = idx >> 3;             // 0..63
            const int c8 = (idx & 7) * 8;        // 0,8,..,56

            // X row-major: vector load + pack + single b128 LDS store.
            const float* xp = X + (by * 64 + r) * D_ + kk + c8;
            v4f x0 = *(const v4f*)(xp);
            v4f x1 = *(const v4f*)(xp + 4);
            v8h xs8;
            #pragma unroll
            for (int i = 0; i < 4; ++i) {
                xs8[i]     = (_Float16)x0[i];
                xs8[i + 4] = (_Float16)x1[i];
            }
            *(v8h*)(&Xs[r][c8]) = xs8;

            // W transposed into WsT: vector global load, b16 scatter.
            const float* wp = W + (kk + r) * D_ + bx * 64 + c8;
            v4f w0 = *(const v4f*)(wp);
            v4f w1 = *(const v4f*)(wp + 4);
            #pragma unroll
            for (int i = 0; i < 4; ++i) {
                WsT[c8 + i][r]     = (_Float16)w0[i];
                WsT[c8 + i + 4][r] = (_Float16)w1[i];
            }
        }
        __syncthreads();

        #pragma unroll
        for (int s = 0; s < 2; ++s) {
            const int st = st0 + s;
            const int tr = st >> 2, tc = st & 3;
            // Transposed product: A rows = output columns, B cols = output rows.
            const _Float16* abase = &WsT[tc * 16 + ln16][0];
            const _Float16* bbase = &Xs[tr * 16 + ln16][0];
            #pragma unroll
            for (int k0 = 0; k0 < 64; k0 += 32) {
                v16h a = load_frag_a(abase + k0, half);
                v16h b = load_frag_b(bbase + k0, half);
                acc[s] = __builtin_amdgcn_wmma_f32_16x16x32_f16(
                    false, a, false, b, (short)0, acc[s], false, false);
            }
        }
        __syncthreads();
    }

    // Epilogue: lane owns row (by*64 + tr*16 + ln16), cols colb..colb+7.
    #pragma unroll
    for (int s = 0; s < 2; ++s) {
        const int st = st0 + s;
        const int tr = st >> 2, tc = st & 3;
        const int rowg = by * 64 + tr * 16 + ln16;
        const int colb = bx * 64 + tc * 16 + half * 8;
        v4f b0 = *(const v4f*)(bias + colb);
        v4f b1 = *(const v4f*)(bias + colb + 4);
        v8h ov;
        #pragma unroll
        for (int i = 0; i < 4; ++i) {
            ov[i]     = (_Float16)(acc[s][i]     + b0[i]);
            ov[i + 4] = (_Float16)(acc[s][i + 4] + b1[i]);
        }
        *(v8h*)(out + rowg * D_ + colb) = ov;   // one b128 store
    }
}

// ---------------------------------------------------------------------------
// Kernel 2: lambda gates (vectorized f16 loads).
// ---------------------------------------------------------------------------
__global__ __launch_bounds__(256)
void lambda_kernel(const _Float16* __restrict__ qh,  const _Float16* __restrict__ kh,
                   const _Float16* __restrict__ qah, const _Float16* __restrict__ kah,
                   const float* __restrict__ wlqc, const float* __restrict__ blqc,
                   const float* __restrict__ wlqq, const float* __restrict__ blqq,
                   const float* __restrict__ wlkc, const float* __restrict__ blkc,
                   const float* __restrict__ wlkk, const float* __restrict__ blkk,
                   float* __restrict__ lamq, float* __restrict__ lamk)
{
    int idx = blockIdx.x * blockDim.x + threadIdx.x;
    if (idx >= B_ * H_ * S_) return;
    int s = idx % S_;
    int h = (idx / S_) % H_;
    int b = idx / (H_ * S_);
    const int base = (b * S_ + s) * D_ + h * DK_;

    float aq = blqc[0] + blqq[0];
    float ak = blkc[0] + blkk[0];
    #pragma unroll
    for (int d8 = 0; d8 < 8; ++d8) {
        v8h qa8 = *(const v8h*)(qah + base + d8 * 8);
        v8h q8  = *(const v8h*)(qh  + base + d8 * 8);
        v8h ka8 = *(const v8h*)(kah + base + d8 * 8);
        v8h k8  = *(const v8h*)(kh  + base + d8 * 8);
        #pragma unroll
        for (int i = 0; i < 8; ++i) {
            int d = d8 * 8 + i;
            aq += (float)qa8[i] * wlqc[d] + (float)q8[i] * wlqq[d];
            ak += (float)ka8[i] * wlkc[d] + (float)k8[i] * wlkk[d];
        }
    }
    lamq[idx] = 1.f / (1.f + expf(-aq));
    lamk[idx] = 1.f / (1.f + expf(-ak));
}

// ---------------------------------------------------------------------------
// Kernel 3: dual-score attention, scores computed TRANSPOSED:
//   scores^T = K_tile @ Q^T  (A = key rows, B = query rows)
// -> each lane owns 8 consecutive key-columns of one query row:
//    vectorized mask/lamk loads and single b128 LDS stores.
// Pass 2: contiguous column ownership, exp cached in registers,
// v8h LDS loads, v4f global stores.
// ---------------------------------------------------------------------------
__global__ __launch_bounds__(128)
void attn_kernel(const _Float16* __restrict__ qh,  const _Float16* __restrict__ kh,
                 const _Float16* __restrict__ qah, const _Float16* __restrict__ kah,
                 const int* __restrict__ mask,
                 const float* __restrict__ lamq, const float* __restrict__ lamk,
                 float* __restrict__ out)
{
    __shared__ _Float16 sc[16 * 1024];   // masked scaled scores
    __shared__ _Float16 qu[16 * 1024];   // lambda_context * sigmoid(aspect)

    const int tid  = threadIdx.x;
    const int lane = tid & 31;
    const int wave = tid >> 5;
    const int half = lane >> 4;
    const int ln16 = lane & 15;
    const int qt = blockIdx.x;           // query tile (0..63)
    const int bh = blockIdx.y;           // b*H + h    (0..47)
    const int b  = bh / H_;
    const int h  = bh % H_;
    const int q0 = qt * 16;
    const float scale = 0.125f;          // 1/sqrt(64)

    // B fragments from the query side (loaded once): lane = query row.
    const int arow = b * S_ * D_ + (q0 + ln16) * D_ + h * DK_;
    v16h bq0 = load_frag_b(qh  + arow, half);
    v16h bq1 = load_frag_b(qh  + arow + 32, half);
    v16h ba0 = load_frag_b(qah + arow, half);
    v16h ba1 = load_frag_b(qah + arow + 32, half);

    const float lq = lamq[bh * S_ + q0 + ln16];            // fixed query row
    const int*   mrow   = mask + (b * S_ + q0 + ln16) * S_;
    const float* lkrow  = lamk + bh * S_;

    for (int t = wave; t < 64; t += 4) {
        const int krow = b * S_ * D_ + (t * 16 + ln16) * D_ + h * DK_;
        v16h ak0 = load_frag_a(kh  + krow, half);
        v16h ak1 = load_frag_a(kh  + krow + 32, half);
        v16h aa0 = load_frag_a(kah + krow, half);
        v16h aa1 = load_frag_a(kah + krow + 32, half);

        v8f cs = {}, ca = {};
        cs = __builtin_amdgcn_wmma_f32_16x16x32_f16(false, ak0, false, bq0, (short)0, cs, false, false);
        cs = __builtin_amdgcn_wmma_f32_16x16x32_f16(false, ak1, false, bq1, (short)0, cs, false, false);
        ca = __builtin_amdgcn_wmma_f32_16x16x32_f16(false, aa0, false, ba0, (short)0, ca, false, false);
        ca = __builtin_amdgcn_wmma_f32_16x16x32_f16(false, aa1, false, ba1, (short)0, ca, false, false);

        const int colb = t * 16 + half * 8;                // 8 consecutive cols
        v4i m0 = *(const v4i*)(mrow + colb);
        v4i m1 = *(const v4i*)(mrow + colb + 4);
        v4f l0 = *(const v4f*)(lkrow + colb);
        v4f l1 = *(const v4f*)(lkrow + colb + 4);

        v8h svec, qvec;
        #pragma unroll
        for (int i = 0; i < 8; ++i) {
            int   mv  = (i < 4) ? m0[i] : m1[i - 4];
            float lk  = (i < 4) ? l0[i] : l1[i - 4];
            float sv   = (mv == 0) ? -30000.f : cs[i] * scale;
            float qa_s = (mv == 0) ? 0.f : 1.f / (1.f + expf(-ca[i] * scale));
            svec[i] = (_Float16)sv;
            qvec[i] = (_Float16)((1.f - (lq + lk)) * qa_s);
        }
        *(v8h*)(&sc[ln16 * 1024 + colb]) = svec;   // one b128 store each
        *(v8h*)(&qu[ln16 * 1024 + colb]) = qvec;
    }
    __syncthreads();

    // Pass 2: row softmax.  8 consecutive lanes per row, each owning a
    // contiguous 128-column block -> all LDS traffic is b128.
    const int row   = tid >> 3;   // 0..15
    const int chunk = tid & 7;    // 0..7
    const _Float16* srow = &sc[row * 1024 + chunk * 128];

    v8h ex[16];
    float rmax = -1e30f;
    #pragma unroll
    for (int j = 0; j < 16; ++j) {
        v8h v = *(const v8h*)(srow + j * 8);
        ex[j] = v;
        #pragma unroll
        for (int i = 0; i < 8; ++i) rmax = fmaxf(rmax, (float)v[i]);
    }
    rmax = fmaxf(rmax, __shfl_xor(rmax, 1));
    rmax = fmaxf(rmax, __shfl_xor(rmax, 2));
    rmax = fmaxf(rmax, __shfl_xor(rmax, 4));

    float rsum = 0.f;
    #pragma unroll
    for (int j = 0; j < 16; ++j) {
        v8h v = ex[j];
        v8h e;
        #pragma unroll
        for (int i = 0; i < 8; ++i) {
            float tv = expf((float)v[i] - rmax);
            rsum += tv;
            e[i] = (_Float16)tv;
        }
        ex[j] = e;                      // exp cached in registers
    }
    rsum += __shfl_xor(rsum, 1);
    rsum += __shfl_xor(rsum, 2);
    rsum += __shfl_xor(rsum, 4);
    const float inv = 1.f / rsum;

    const _Float16* qrow = &qu[row * 1024 + chunk * 128];
    float* orow = out + ((size_t)(bh * S_ + q0 + row)) * S_ + chunk * 128;
    #pragma unroll
    for (int j = 0; j < 16; ++j) {
        v8h e  = ex[j];
        v8h qv = *(const v8h*)(qrow + j * 8);
        v4f o0, o1;
        #pragma unroll
        for (int i = 0; i < 4; ++i) {
            o0[i] = (float)e[i]     * inv + (float)qv[i];
            o1[i] = (float)e[i + 4] * inv + (float)qv[i + 4];
        }
        *(v4f*)(orow + j * 8)     = o0;   // b128 stores
        *(v4f*)(orow + j * 8 + 4) = o1;
    }
}

// ---------------------------------------------------------------------------
extern "C" void kernel_launch(void* const* d_in, const int* in_sizes, int n_in,
                              void* d_out, int out_size, void* d_ws, size_t ws_size,
                              hipStream_t stream)
{
    (void)in_sizes; (void)n_in; (void)out_size; (void)ws_size;
    const float* query  = (const float*)d_in[0];
    const float* key    = (const float*)d_in[1];
    const float* aspect = (const float*)d_in[2];
    const int*   mask   = (const int*)d_in[3];
    const float* Wq  = (const float*)d_in[4];  const float* bq  = (const float*)d_in[5];
    const float* Wk  = (const float*)d_in[6];  const float* bk  = (const float*)d_in[7];
    const float* Wqa = (const float*)d_in[8];  const float* bqa = (const float*)d_in[9];
    const float* Wka = (const float*)d_in[10]; const float* bka = (const float*)d_in[11];
    const float* wlqc = (const float*)d_in[12]; const float* blqc = (const float*)d_in[13];
    const float* wlqq = (const float*)d_in[14]; const float* blqq = (const float*)d_in[15];
    const float* wlkc = (const float*)d_in[16]; const float* blkc = (const float*)d_in[17];
    const float* wlkk = (const float*)d_in[18]; const float* blkk = (const float*)d_in[19];
    float* out = (float*)d_out;

    char* ws = (char*)d_ws;
    const size_t N1 = (size_t)B_ * S_ * D_;            // 3,145,728 elems
    _Float16* qh  = (_Float16*)(ws);
    _Float16* kh  = (_Float16*)(ws + 2 * N1);
    _Float16* qah = (_Float16*)(ws + 4 * N1);
    _Float16* kah = (_Float16*)(ws + 6 * N1);
    float* lamq = (float*)(ws + 8 * N1);
    float* lamk = (float*)(ws + 8 * N1 + (size_t)B_ * H_ * S_ * sizeof(float));

    dim3 gA(D_ / 64, (B_ * S_) / 64, 4);
    proj_kernel<<<gA, 256, 0, stream>>>(query, key, aspect,
                                        Wq, bq, Wk, bk, Wqa, bqa, Wka, bka,
                                        qh, kh, qah, kah);

    const int nl = B_ * H_ * S_;
    lambda_kernel<<<(nl + 255) / 256, 256, 0, stream>>>(qh, kh, qah, kah,
                                                        wlqc, blqc, wlqq, blqq,
                                                        wlkc, blkc, wlkk, blkk,
                                                        lamq, lamk);

    dim3 gC(S_ / 16, B_ * H_);
    attn_kernel<<<gC, 128, 0, stream>>>(qh, kh, qah, kah, mask, lamq, lamk, out);
}
